// DGN5_70428873720432
// MI455X (gfx1250) — matrix operand order repaired and checked by reading, not compiled, not run on hardware
//
#include <hip/hip_runtime.h>
#include <math.h>

// ---------------------------------------------------------------------------
// Problem constants (from reference: B=4, T=2048, D=1024, K=8)
// ---------------------------------------------------------------------------
constexpr int Bc   = 4;
constexpr int Tc   = 2048;
constexpr int Dc   = 1024;
constexpr int KTOP = 8;
constexpr int TILE = 16;
constexpr int NW   = 4;               // waves per block (share one A tile)
constexpr int NKSTEP = Dc / 32;       // 32 wmma K-steps per 16x16 score tile
constexpr int LDA  = Dc + 8;          // padded bf16 elems per LDS row (16B pad)

typedef __attribute__((ext_vector_type(16))) __bf16 bf16x16;
typedef __attribute__((ext_vector_type(8)))  __bf16 bf16x8;
typedef __attribute__((ext_vector_type(8)))  float  f32x8;

// ---------------------------------------------------------------------------
// Phase 1: fp32 -> bf16 (RNE via bit trick), 4 elements / thread
// ---------------------------------------------------------------------------
static __device__ __forceinline__ unsigned int f2bf(float f) {
  unsigned int u = __float_as_uint(f);
  return (u + 0x7fffu + ((u >> 16) & 1u)) >> 16;
}

__global__ void cvt_bf16_kernel(const float* __restrict__ x,
                                unsigned long long* __restrict__ xb,
                                int n4) {
  int i = blockIdx.x * blockDim.x + threadIdx.x;
  if (i < n4) {
    float4 v = ((const float4*)x)[i];
    unsigned long long p =  (unsigned long long)f2bf(v.x)
                         | ((unsigned long long)f2bf(v.y) << 16)
                         | ((unsigned long long)f2bf(v.z) << 32)
                         | ((unsigned long long)f2bf(v.w) << 48);
    xb[i] = p;
  }
}

// ---------------------------------------------------------------------------
// Phase 2: fused scores(WMMA bf16) -> causal top-8 -> gather-mean -> GELU.
// 4 waves / block share one staged 16-row A tile; wave w takes column tiles
// jt = w, w+NW, ...  Intra-wave LDS transpose ordered by wavefront fence.
// ---------------------------------------------------------------------------
__global__ void __launch_bounds__(NW * 32)
graph_topk_fused_kernel(const float* __restrict__ x,      // (B,T,D) f32
                        const __bf16* __restrict__ xb,    // (B,T,D) bf16
                        const float* __restrict__ gain,   // (D,)
                        const float* __restrict__ bias,   // (D,)
                        const float* __restrict__ log_mix,
                        const float* __restrict__ log_scale,
                        float* __restrict__ out)          // (B,T,D) f32
{
  __shared__ __align__(16) __bf16 sA[TILE * LDA];     // staged A tile (33 KB)
  __shared__ float sS[NW * TILE * TILE];              // per-wave 16x16 transpose
  __shared__ float sMV[TILE * NW * TILE];             // merge: 16 rows x 64 vals
  __shared__ int   sMI[TILE * NW * TILE];             // merge: 16 rows x 64 idx
  __shared__ int   sIdx[TILE * KTOP];                 // final neighbor lists
  __shared__ int   sDeg[TILE];

  const int tid  = threadIdx.x;
  const int wave = tid >> 5;
  const int lane = tid & 31;
  const int half = lane >> 4;            // 0 or 1
  const int mrow = lane & 15;            // row within tile (and col id for B)
  const int ntile = Tc / TILE;           // 128
  const int b  = blockIdx.x / ntile;
  const int it = (ntile - 1) - (blockIdx.x % ntile);  // heavy tiles first
  const int row0 = it * TILE;

  const __bf16* xbB = xb + (size_t)b * Tc * Dc;
  const float*  xB  = x  + (size_t)b * Tc * Dc;

  // ---- stage A tile (16 rows x 1024 bf16) into padded LDS (all 128 thr) --
  {
    const uint4* src = (const uint4*)(xbB + (size_t)row0 * Dc); // rows contiguous
    for (int c = tid; c < TILE * (Dc / 8); c += NW * 32) {
      int r = c >> 7, q = c & 127;                  // Dc/8 = 128 uint4 per row
      *(uint4*)(&sA[r * LDA + q * 8]) = src[c];
    }
  }
  __syncthreads();

  // ---- private top-8 state (min-replace, fully unrolled) -----------------
  float tv[KTOP]; int ti[KTOP];
  float vmin = -INFINITY; int mslot = 0;
#pragma unroll
  for (int s = 0; s < KTOP; ++s) { tv[s] = -INFINITY; ti[s] = -1; }

  auto insert = [&](float val, int j) {
    if (val > vmin) {
#pragma unroll
      for (int s = 0; s < KTOP; ++s)
        if (s == mslot) { tv[s] = val; ti[s] = j; }
      vmin = tv[0]; mslot = 0;
#pragma unroll
      for (int s = 1; s < KTOP; ++s)
        if (tv[s] < vmin) { vmin = tv[s]; mslot = s; }
    }
  };

  const int i_global = row0 + mrow;
  const __bf16* aptr = &sA[mrow * LDA];
  float* mySS = &sS[wave * TILE * TILE];

  // ---- causal column-tile loop (strided across the 4 waves) --------------
  for (int jt = wave; jt <= it; jt += NW) {
    const __bf16* bptr = xbB + (size_t)(jt * TILE + mrow) * Dc;
    f32x8 acc = {};
#pragma unroll 4
    for (int kk = 0; kk < NKSTEP; ++kk) {
      const int k0 = kk * 32;
      // A fragment (16-bit A 16x32 layout): K = [k0+8h..+7] U [k0+16+8h..+7]
      bf16x8 a_lo = *(const bf16x8*)(aptr + k0 + 8 * half);
      bf16x8 a_hi = *(const bf16x8*)(aptr + k0 + 16 + 8 * half);
      bf16x16 afrag = __builtin_shufflevector(a_lo, a_hi,
          0,1,2,3,4,5,6,7,8,9,10,11,12,13,14,15);
      // B fragment (32x16, same K order as A): K = [k0+8h..+7] U [k0+16+8h..+7]
      bf16x8 b_lo = *(const bf16x8*)(bptr + k0 + 8 * half);
      bf16x8 b_hi = *(const bf16x8*)(bptr + k0 + 16 + 8 * half);
      bf16x16 bfrag = __builtin_shufflevector(b_lo, b_hi,
          0,1,2,3,4,5,6,7,8,9,10,11,12,13,14,15);
      acc = __builtin_amdgcn_wmma_f32_16x16x32_bf16(
          false, afrag, false, bfrag, (short)0, acc, false, false);
    }
    // intra-wave 16x16 transpose through private LDS buffer:
    // acc[r] = S[r + 8*half][mrow]  ->  lane reads S[mrow][8*half + c]
    __builtin_amdgcn_fence(__ATOMIC_ACQ_REL, "wavefront");
    __builtin_amdgcn_wave_barrier();
#pragma unroll
    for (int r = 0; r < 8; ++r)
      mySS[(r + 8 * half) * TILE + mrow] = acc[r];
    __builtin_amdgcn_fence(__ATOMIC_ACQ_REL, "wavefront");
    __builtin_amdgcn_wave_barrier();
#pragma unroll
    for (int c = 0; c < 8; ++c) {
      int n = 8 * half + c;
      int j = jt * TILE + n;
      float val = mySS[mrow * TILE + n];
      if (j <= i_global) insert(val, j);
    }
  }

  // ---- merge the NW*2 = 8 partial top-8 lists per row --------------------
  __syncthreads();
#pragma unroll
  for (int s = 0; s < KTOP; ++s) {
    int col = wave * TILE + half * 8 + s;
    sMV[mrow * (NW * TILE) + col] = tv[s];
    sMI[mrow * (NW * TILE) + col] = ti[s];
  }
  __syncthreads();
  if (tid < TILE) {
    vmin = -INFINITY; mslot = 0;
#pragma unroll
    for (int s = 0; s < KTOP; ++s) { tv[s] = -INFINITY; ti[s] = -1; }
    for (int c = 0; c < NW * TILE; ++c) {
      float val = sMV[tid * (NW * TILE) + c];
      int   j   = sMI[tid * (NW * TILE) + c];
      if (j >= 0) insert(val, j);
    }
    int deg = 0;
#pragma unroll
    for (int s = 0; s < KTOP; ++s) {
      sIdx[tid * KTOP + s] = ti[s];
      deg += (ti[s] >= 0) ? 1 : 0;
    }
    sDeg[tid] = deg;
  }
  __syncthreads();

  // ---- gather-mean + blend + gelu + scale (fp32 path; 4 rows per wave) ---
  const float lmix  = log_mix[0];
  const float mix   = 1.0f / (1.0f + __expf(-lmix));
  const float ls    = log_scale[0];
  const float sp    = (ls > 20.0f) ? ls : log1pf(__expf(ls));
  const float scale = sp + 0.01f;
  const float omix  = 1.0f - mix;
  const int dbase = lane * 4;

  for (int mi = 0; mi < TILE / NW; ++mi) {
    const int m = wave + NW * mi;
    const int i = row0 + m;
    int nb[KTOP];
#pragma unroll
    for (int s = 0; s < KTOP; ++s) nb[s] = sIdx[m * KTOP + s];
    int degi = sDeg[m]; if (degi < 1) degi = 1;
    const float invdeg = 1.0f / (float)degi;
    const float* xi = xB + (size_t)i * Dc;
    float* outp = out + ((size_t)b * Tc + i) * Dc;
    for (int t = 0; t < Dc / 128; ++t) {
      const int d = dbase + t * 128;
      float4 sum = {0.f, 0.f, 0.f, 0.f};
#pragma unroll
      for (int s = 0; s < KTOP; ++s) {
        int j = nb[s];
        if (j >= 0) {
          float4 v = *(const float4*)(xB + (size_t)j * Dc + d);
          sum.x += v.x; sum.y += v.y; sum.z += v.z; sum.w += v.w;
        }
      }
      float4 xv = *(const float4*)(xi + d);
      float4 g  = *(const float4*)(gain + d);
      float4 bi = *(const float4*)(bias + d);
      float4 o;
      {
        float p0 = (mix * xv.x + omix * (sum.x * invdeg)) * g.x + bi.x;
        float p1 = (mix * xv.y + omix * (sum.y * invdeg)) * g.y + bi.y;
        float p2 = (mix * xv.z + omix * (sum.z * invdeg)) * g.z + bi.z;
        float p3 = (mix * xv.w + omix * (sum.w * invdeg)) * g.w + bi.w;
        o.x = 0.5f * p0 * (1.0f + erff(p0 * 0.70710678118654752f)) * scale;
        o.y = 0.5f * p1 * (1.0f + erff(p1 * 0.70710678118654752f)) * scale;
        o.z = 0.5f * p2 * (1.0f + erff(p2 * 0.70710678118654752f)) * scale;
        o.w = 0.5f * p3 * (1.0f + erff(p3 * 0.70710678118654752f)) * scale;
      }
      *(float4*)(outp + d) = o;
    }
  }
}

// ---------------------------------------------------------------------------
extern "C" void kernel_launch(void* const* d_in, const int* in_sizes, int n_in,
                              void* d_out, int out_size, void* d_ws, size_t ws_size,
                              hipStream_t stream) {
  const float* x         = (const float*)d_in[0];
  const float* gain      = (const float*)d_in[1];
  const float* bias      = (const float*)d_in[2];
  const float* log_mix   = (const float*)d_in[3];
  const float* log_scale = (const float*)d_in[4];
  float* out = (float*)d_out;

  // Phase 1: x (fp32) -> bf16 workspace (16.8 MB; fits the 192 MB L2)
  const int n4 = (Bc * Tc * Dc) / 4;
  cvt_bf16_kernel<<<(n4 + 255) / 256, 256, 0, stream>>>(
      x, (unsigned long long*)d_ws, n4);

  // Phase 2: fused scores/top-k/aggregate/GELU; 4 waves share one row tile
  dim3 grid(Bc * (Tc / TILE));
  graph_topk_fused_kernel<<<grid, NW * 32, 0, stream>>>(
      x, (const __bf16*)d_ws, gain, bias, log_mix, log_scale, out);
}
